// Model_57758720197173
// MI455X (gfx1250) — compile-verified, matrix operand order checked by reading, not run on hardware
//
#include <hip/hip_runtime.h>
#include <math.h>

// ---------------- problem constants (match reference) ----------------
#define N_NODES  100000
#define N_EDGES  1600000
#define IN_C     512
#define HID_C    128
#define OUT_C    64
#define N_LAYERS 8
#define ALPHA    0.1f

typedef __attribute__((ext_vector_type(16))) __bf16 v16bf;
typedef __attribute__((ext_vector_type(8)))  __bf16 v8bf;
typedef __attribute__((ext_vector_type(8)))  float  v8f;

static __device__ __forceinline__ __bf16 f2bf(float f) {
    union { float f; unsigned u; } in; in.f = f;
    unsigned u = in.u;
    u += 0x7FFFu + ((u >> 16) & 1u);          // round-to-nearest-even
    union { unsigned short s; __bf16 b; } out;
    out.s = (unsigned short)(u >> 16);
    return out.b;
}

// Guaranteed-native L2 float atomic (returnless form -> TH[0]=0, no VGPR ret).
// Avoids any chance of clang expanding atomicrmw fadd into a CAS loop.
static __device__ __forceinline__ void atomic_add_f32(float* p, float v) {
    asm volatile("global_atomic_add_f32 %0, %1, off"
                 :
                 : "v"(p), "v"(v)
                 : "memory");
}

// ---------------- weight conversion: f32 -> bf16 (+ transpose W_convs) ----
__global__ void prep_weights(const float* __restrict__ Wenc,
                             const float* __restrict__ Wconv,
                             const float* __restrict__ Wdec,
                             __bf16* __restrict__ WencB,
                             __bf16* __restrict__ WconvT,
                             __bf16* __restrict__ WdecB) {
    int gid = blockIdx.x * blockDim.x + threadIdx.x;
    const int nEnc  = HID_C * IN_C;                 // 65536
    const int nConv = N_LAYERS * HID_C * HID_C;     // 131072
    const int nDec  = OUT_C * HID_C;                // 8192
    if (gid < nEnc) { WencB[gid] = f2bf(Wenc[gid]); return; }
    gid -= nEnc;
    if (gid < nConv) {
        int l = gid / (HID_C * HID_C);
        int r = gid % (HID_C * HID_C);
        int n = r / HID_C, k = r % HID_C;
        // B for hh @ W is W itself; store transposed so B columns are contiguous
        WconvT[(size_t)(l * HID_C + n) * HID_C + k] =
            f2bf(Wconv[(size_t)(l * HID_C + k) * HID_C + n]);
        return;
    }
    gid -= nConv;
    if (gid < nDec) WdecB[gid] = f2bf(Wdec[gid]);
}

// ---------------- agg' = (a/(1-a)) * x0  (replaces zero-fill) ------------
__global__ void init_agg(float4* __restrict__ agg, const float4* __restrict__ x0,
                         long n4) {
    long gid = (long)blockIdx.x * blockDim.x + threadIdx.x;
    if (gid >= n4) return;
    const float s = ALPHA / (1.f - ALPHA);
    float4 z = x0[gid];
    agg[gid] = make_float4(s * z.x, s * z.y, s * z.z, s * z.w);
}

// ---------------- packed f32 -> bf16 conversion pass ---------------------
__global__ void cvt_bf16(const float4* __restrict__ in, uint2* __restrict__ out,
                         long n4) {
    long gid = (long)blockIdx.x * blockDim.x + threadIdx.x;
    if (gid >= n4) return;
    float4 v = in[gid];
    union { __bf16 b[4]; uint2 u; } o;
    o.b[0] = f2bf(v.x); o.b[1] = f2bf(v.y); o.b[2] = f2bf(v.z); o.b[3] = f2bf(v.w);
    out[gid] = o.u;
}

// ---------------- SpMM: agg[d] += w_e * h[src_e]  (edge-parallel) --------
// one wave32 per edge, each lane handles 4 channels (float4); both the
// gathered h rows and the atomic targets are L2-resident (51 MB < 192 MB L2)
__global__ void spmm_agg(const int*   __restrict__ src,
                         const int*   __restrict__ dst,
                         const float* __restrict__ ew,
                         const float* __restrict__ h,
                         float*       __restrict__ agg) {
    long gid = (long)blockIdx.x * blockDim.x + threadIdx.x;
    int e = (int)(gid >> 5);
    if (e >= N_EDGES) return;
    int c = ((int)gid & 31) * 4;
    int s = src[e], d = dst[e];
    float w = ew[e];
    const float4 hv = *(const float4*)(h + (size_t)s * HID_C + c);
    float* out = agg + (size_t)d * HID_C + c;
    atomic_add_f32(out + 0, w * hv.x);
    atomic_add_f32(out + 1, w * hv.y);
    atomic_add_f32(out + 2, w * hv.z);
    atomic_add_f32(out + 3, w * hv.w);
}

// ---------------- WMMA GEMM: D = epilogue(A[M,K] @ B^T), B given as [N,K] --
// Fully specialized at compile time; wave w of the block owns the 16x16
// output tile at (blockIdx.x*16, w*16). A operand striping per CDNA5 ISA:
// lanes 0-15 -> row M=lane%16, K chunks {0..7,16..23}; lanes 16-31 -> K
// chunks {8..15,24..31}. B column n == row n of the [N,K] array (contiguous).
template<int K, int N, bool AF32, bool HASC, bool RELU, bool DBF, bool D2F>
__global__ __launch_bounds__(256)
void wmma_gemm(const float*  __restrict__ Af,   // f32 A (fallback path)
               const __bf16* __restrict__ Abf,  // bf16 A
               const __bf16* __restrict__ B,    // [N,K] row-major
               const float*  __restrict__ Cin,  // epilogue linear term
               float*        __restrict__ D,
               __bf16*       __restrict__ Dbf,  // optional bf16 mirror
               float*        __restrict__ D2,   // optional extra f32 copy
               float wLin, float wAcc) {
    const int lane  = threadIdx.x & 31;
    const int wave  = threadIdx.x >> 5;
    const int mtile = blockIdx.x * 16;
    const int ntile = wave * 16;
    const int arow  = lane & 15;
    const int koff  = (lane >> 4) * 8;
    const int bcol  = ntile + arow;

    v8f acc = {};
    #pragma unroll
    for (int k0 = 0; k0 < K; k0 += 32) {
        v16bf a, b;
        if constexpr (AF32) {
            const float* ap = Af + (size_t)(mtile + arow) * K + koff + k0;
            #pragma unroll
            for (int i = 0; i < 8; ++i) { a[i] = f2bf(ap[i]); a[8 + i] = f2bf(ap[16 + i]); }
        } else {
            const v8bf* ap = (const v8bf*)(Abf + (size_t)(mtile + arow) * K + koff + k0);
            v8bf alo = ap[0];
            v8bf ahi = ap[2];                   // +16 halfs
            #pragma unroll
            for (int i = 0; i < 8; ++i) { a[i] = alo[i]; a[8 + i] = ahi[i]; }
        }
        {
            const v8bf* bp = (const v8bf*)(B + (size_t)bcol * K + koff + k0);
            v8bf blo = bp[0];
            v8bf bhi = bp[2];
            #pragma unroll
            for (int i = 0; i < 8; ++i) { b[i] = blo[i]; b[8 + i] = bhi[i]; }
        }
        acc = __builtin_amdgcn_wmma_f32_16x16x32_bf16(
                  /*neg_a=*/false, a, /*neg_b=*/false, b,
                  /*c_mod=*/(short)0, acc, /*reuse_a=*/false, /*reuse_b=*/false);
    }

    // C/D layout: VGPR r, lanes 0-15 -> M=r, lanes 16-31 -> M=r+8, N=lane%16
    const int col   = ntile + arow;
    const int rbase = (lane < 16) ? 0 : 8;
    #pragma unroll
    for (int r = 0; r < 8; ++r) {
        const size_t idx = (size_t)(mtile + rbase + r) * N + col;
        float v = wAcc * acc[r];
        if constexpr (HASC) v += wLin * Cin[idx];
        if constexpr (RELU) v = v > 0.f ? v : 0.f;
        D[idx] = v;
        if constexpr (DBF) Dbf[idx] = f2bf(v);
        if constexpr (D2F) D2[idx]  = v;
    }
}

// ---------------------------- host driver --------------------------------
extern "C" void kernel_launch(void* const* d_in, const int* in_sizes, int n_in,
                              void* d_out, int out_size, void* d_ws, size_t ws_size,
                              hipStream_t stream) {
    const float* x     = (const float*)d_in[0];
    const int*   eidx  = (const int*)d_in[1];
    const float* ew    = (const float*)d_in[2];
    const float* Wenc  = (const float*)d_in[3];
    const float* Wconv = (const float*)d_in[4];
    const float* Wdec  = (const float*)d_in[5];
    const int* src = eidx;
    const int* dst = eidx + N_EDGES;

    char* ws  = (char*)d_ws;
    char* end = (char*)d_ws + ws_size;
    auto carve = [&](size_t bytes) -> char* {
        char* p = ws;
        ws += (bytes + 255) & ~(size_t)255;
        return p;
    };
    float*  h      = (float*)carve((size_t)N_NODES * HID_C * 4);
    float*  x0     = (float*)carve((size_t)N_NODES * HID_C * 4);
    float*  agg    = (float*)carve((size_t)N_NODES * HID_C * 4);   // agg' = hh/(1-a)
    __bf16* hbf    = (__bf16*)carve((size_t)N_NODES * HID_C * 2);
    __bf16* aggbf  = (__bf16*)carve((size_t)N_NODES * HID_C * 2);
    __bf16* WencB  = (__bf16*)carve((size_t)HID_C * IN_C * 2);
    __bf16* WconvT = (__bf16*)carve((size_t)N_LAYERS * HID_C * HID_C * 2);
    __bf16* WdecB  = (__bf16*)carve((size_t)OUT_C * HID_C * 2);
    // optional bf16 mirror of x (pre-converted encoder input)
    const size_t xbfBytes = (size_t)N_NODES * IN_C * 2;
    bool preCvtX = (size_t)(end - ws) >= xbfBytes;
    __bf16* xbf = preCvtX ? (__bf16*)carve(xbfBytes) : nullptr;

    const int totW = HID_C * IN_C + N_LAYERS * HID_C * HID_C + OUT_C * HID_C;
    prep_weights<<<(totW + 255) / 256, 256, 0, stream>>>(Wenc, Wconv, Wdec,
                                                         WencB, WconvT, WdecB);

    const int  mBlocks = N_NODES / 16;                      // 6250
    const long n4      = (long)N_NODES * HID_C / 4;         // 3.2M float4
    const int  n4Blk   = (int)((n4 + 255) / 256);
    const int  eBlk    = (int)(((long)N_EDGES * 32 + 255) / 256);

    // encoder: h = x @ Wenc^T  (also x0 = h, hbf = bf16(h))
    if (preCvtX) {
        const long xn4 = (long)N_NODES * IN_C / 4;
        cvt_bf16<<<(int)((xn4 + 255) / 256), 256, 0, stream>>>(
            (const float4*)x, (uint2*)xbf, xn4);
        wmma_gemm<IN_C, HID_C, false, false, false, true, true>
            <<<mBlocks, 256, 0, stream>>>(nullptr, xbf, WencB, nullptr,
                                          h, hbf, x0, 0.f, 1.f);
    } else {
        wmma_gemm<IN_C, HID_C, true, false, false, true, true>
            <<<mBlocks, 256, 0, stream>>>(x, nullptr, WencB, nullptr,
                                          h, hbf, x0, 0.f, 1.f);
    }

    for (int i = 0; i < N_LAYERS; ++i) {
        // agg' = (a/(1-a))*x0; after atomics, hh = (1-a)*agg'
        init_agg<<<n4Blk, 256, 0, stream>>>((float4*)agg, (const float4*)x0, n4);
        spmm_agg<<<eBlk, 256, 0, stream>>>(src, dst, ew, h, agg);
        cvt_bf16<<<n4Blk, 256, 0, stream>>>((const float4*)agg, (uint2*)aggbf, n4);
        float beta = logf(0.5f / (float)(i + 1) + 1.0f);
        // h = relu((1-b)(1-a)*agg' + b(1-a)*(agg' @ Wconv_i))
        wmma_gemm<HID_C, HID_C, false, true, true, true, false>
            <<<mBlocks, 256, 0, stream>>>(nullptr, aggbf,
                                          WconvT + (size_t)i * HID_C * HID_C,
                                          agg, h, hbf, nullptr,
                                          (1.f - beta) * (1.f - ALPHA),
                                          beta * (1.f - ALPHA));
    }

    // decoder: out = h @ Wdec^T  (N=64 -> 4 waves per block)
    wmma_gemm<HID_C, OUT_C, false, false, false, false, false>
        <<<mBlocks, 128, 0, stream>>>(nullptr, hbf, WdecB, nullptr,
                                      (float*)d_out, nullptr, nullptr, 0.f, 1.f);
}
